// MultiHeadAttention_2164663517677
// MI455X (gfx1250) — matrix operand order repaired and checked by reading, not compiled, not run on hardware
//
#include <hip/hip_runtime.h>

// MHA forward, MI455X (gfx1250): bf16 WMMA + TDM (tensor_load_to_lds) staging.
// d_out = [ out (B,T,C) | k (B,H,T,hd) | v (B,H,T,hd) ], fp32.
// d_ws: bf16 packed copies: Xbf, Wqkv^T, Wproj^T, Qbf, Kbf, V^Tbf, Obf.

#define B_   4
#define T_   2048
#define C_   1024
#define H_   16
#define HD_  64
#define M_   (B_ * T_)      // 8192
#define NQKV 3072

typedef __attribute__((ext_vector_type(16))) __bf16 v16bf;
typedef __attribute__((ext_vector_type(4)))  __bf16 bf16x4;
typedef __attribute__((ext_vector_type(8)))  float  v8f;
typedef __attribute__((ext_vector_type(4)))  unsigned int u32x4;
typedef __attribute__((ext_vector_type(8)))  int i32x8;
typedef __attribute__((ext_vector_type(4)))  int i32x4;

__device__ __forceinline__ __bf16 f2bf(float f) {
  unsigned u = __float_as_uint(f);
  u += 0x7FFFu + ((u >> 16) & 1u);          // round-to-nearest-even
  unsigned short h = (unsigned short)(u >> 16);
  __bf16 r;
  __builtin_memcpy(&r, &h, 2);
  return r;
}

// Generic-pointer -> LDS byte offset (addrspacecast to AS3, then ptrtoint).
typedef __attribute__((address_space(3))) char lds_char_t;
__device__ __forceinline__ unsigned lds_off(const void* p) {
  return (unsigned)(unsigned long long)(lds_char_t*)(void*)(const_cast<void*>(p));
}

// ---------------------------------------------------------------------------
// Tensor Data Mover: 2D bf16 tile (tile_x elems wide, tile_rows rows) from
// global (row stride in elements) into LDS at lds_byte, with 1-DWORD padding
// every 2^(pad_log+1) DWORDs (pad_log=3 -> LD=34 bf16; pad_log=4 -> LD=66).
// ---------------------------------------------------------------------------
__device__ __forceinline__ void tdm_load_2d(unsigned lds_byte, const void* gaddr,
                                            unsigned tile_x, unsigned tile_rows,
                                            unsigned long long row_stride_elems,
                                            unsigned pad_log) {
  const unsigned long long ga = (unsigned long long)gaddr;
  u32x4 g0;
  g0[0] = 1u;                                              // count=1, user desc
  g0[1] = lds_byte;                                        // lds_addr [63:32]
  g0[2] = (unsigned)ga;                                    // global_addr lo
  g0[3] = (unsigned)((ga >> 32) & 0x01FFFFFFu) | 0x80000000u;  // ga hi | type=2
  i32x8 g1;
  g1[0] = (int)((1u << 16) | (1u << 20) | (pad_log << 22)); // data_size=2B, pad_en, pad=1 dword
  g1[1] = 0;                                // tensor_dim0 = 1<<20: lo16=0
  g1[2] = 0x10;                             // dim0 hi16=16 ; tensor_dim1 lo16=0
  g1[3] = (int)(0x10u | (tile_x << 16));    // dim1 hi16=16 ; tile_dim0
  g1[4] = (int)(tile_rows & 0xFFFFu);       // tile_dim1 ; tile_dim2=0
  g1[5] = (int)(unsigned)(row_stride_elems & 0xFFFFFFFFu); // dim0_stride lo32
  g1[6] = (int)(unsigned)((row_stride_elems >> 32) & 0xFFFFu);
  g1[7] = 0;
  const i32x4 z4 = {0, 0, 0, 0};
#if defined(__clang_major__) && (__clang_major__ >= 23)
  const i32x8 z8 = {0, 0, 0, 0, 0, 0, 0, 0};
  __builtin_amdgcn_tensor_load_to_lds(g0, g1, z4, z4, z8, 0);
#else
  __builtin_amdgcn_tensor_load_to_lds(g0, g1, z4, z4, 0);
#endif
}

// A fragment, 16x32 bf16, row-major LDS tile.
__device__ __forceinline__ v16bf frag_a16x32(const __bf16* lds, int row, int kbase, int ld) {
  const int lane = threadIdx.x & 31;
  const __bf16* p = lds + (row + (lane & 15)) * ld + kbase + ((lane >> 4) << 3);
  v16bf f;
#pragma unroll
  for (int v = 0; v < 8; ++v) {
    const int k = ((v & 4) << 2) | ((v & 3) << 1);
    f[2 * v]     = p[k];
    f[2 * v + 1] = p[k + 1];
  }
  return f;
}

// B fragment, 32x16 bf16, from an N-major (transposed) LDS tile.
__device__ __forceinline__ v16bf frag_b32x16(const __bf16* lds, int col, int kbase, int ld) {
  const int lane = threadIdx.x & 31;
  const __bf16* p = lds + (col + (lane & 15)) * ld + kbase + ((lane >> 4) << 3);
  v16bf f;
#pragma unroll
  for (int e = 0; e < 16; ++e) f[e] = p[(e & 7) + ((e >> 3) << 4)];
  return f;
}

__device__ __forceinline__ v8f wmma_bf16(v16bf a, v16bf b, v8f c) {
  return __builtin_amdgcn_wmma_f32_16x16x32_bf16(false, a, false, b, (short)0, c, false, false);
}

// ---------------------------------------------------------------------------
// Pack kernels (one-time, bandwidth-trivial).
// ---------------------------------------------------------------------------
__global__ __launch_bounds__(256) void pack_bf16_k(const float* __restrict__ src,
                                                   __bf16* __restrict__ dst, int n4) {
  const int i = blockIdx.x * 256 + threadIdx.x;
  if (i < n4) {
    const float4 v = ((const float4*)src)[i];
    bf16x4 o;
    o[0] = f2bf(v.x); o[1] = f2bf(v.y); o[2] = f2bf(v.z); o[3] = f2bf(v.w);
    ((bf16x4*)dst)[i] = o;
  }
}

// Wt[n][k] = bf16(W[k][n]); reads coalesced along n.
__global__ __launch_bounds__(256) void transpose_pack_k(const float* __restrict__ W,
                                                        __bf16* __restrict__ Wt,
                                                        int K, int N) {
  const int e = blockIdx.x * 256 + threadIdx.x;
  if (e < K * N) {
    const int n = e % N, k = e / N;
    Wt[(size_t)n * K + k] = f2bf(W[e]);
  }
}

// ---------------------------------------------------------------------------
// Kernel 1: QKV GEMM, k-step 64, double-buffered TDM staging.
// Epilogue: fp32 K,V -> d_out; bf16 Q,K -> [bh][T][hd]; bf16 V^T -> [bh][hd][T].
// ---------------------------------------------------------------------------
__global__ __launch_bounds__(256) void qkv_gemm_tdm(
    const __bf16* __restrict__ Abf, const __bf16* __restrict__ Btbf,
    float* __restrict__ Kout, float* __restrict__ Vout,
    __bf16* __restrict__ Qbf, __bf16* __restrict__ Kbf, __bf16* __restrict__ Vtbf) {
  constexpr int LD = 66;
  __shared__ __bf16 As[2][128 * LD];
  __shared__ __bf16 Bts[2][128 * LD];
  const int tid = threadIdx.x, lane = tid & 31, w = tid >> 5;
  const int m0 = blockIdx.y * 128, n0 = blockIdx.x * 128;
  const int wr = (w & 3) * 32, wc = (w >> 2) * 64;
  const v8f vz = {0.f, 0.f, 0.f, 0.f, 0.f, 0.f, 0.f, 0.f};
  v8f acc[2][4];
#pragma unroll
  for (int mt = 0; mt < 2; ++mt)
#pragma unroll
    for (int nt = 0; nt < 4; ++nt) acc[mt][nt] = vz;

  if (w == 0) {
    tdm_load_2d(lds_off(&As[0][0]),  Abf  + (size_t)m0 * C_, 64, 128, C_, 4);
    tdm_load_2d(lds_off(&Bts[0][0]), Btbf + (size_t)n0 * C_, 64, 128, C_, 4);
  }
  for (int it = 0; it < 16; ++it) {
    const int cur = it & 1;
    if (it + 1 < 16) {
      if (w == 0) {
        tdm_load_2d(lds_off(&As[cur ^ 1][0]),  Abf  + (size_t)m0 * C_ + (it + 1) * 64, 64, 128, C_, 4);
        tdm_load_2d(lds_off(&Bts[cur ^ 1][0]), Btbf + (size_t)n0 * C_ + (it + 1) * 64, 64, 128, C_, 4);
      }
      __builtin_amdgcn_s_wait_tensorcnt(2);   // in-order: current pair done
    } else {
      __builtin_amdgcn_s_wait_tensorcnt(0);
    }
    __syncthreads();
#pragma unroll
    for (int kc = 0; kc < 2; ++kc) {
      v16bf a0 = frag_a16x32(&As[cur][0], wr, kc * 32, LD);
      v16bf a1 = frag_a16x32(&As[cur][0], wr + 16, kc * 32, LD);
#pragma unroll
      for (int nt = 0; nt < 4; ++nt) {
        v16bf bfr = frag_b32x16(&Bts[cur][0], wc + nt * 16, kc * 32, LD);
        acc[0][nt] = wmma_bf16(a0, bfr, acc[0][nt]);
        acc[1][nt] = wmma_bf16(a1, bfr, acc[1][nt]);
      }
    }
    __syncthreads();   // all waves done with this buffer before TDM overwrites
  }

#pragma unroll
  for (int mt = 0; mt < 2; ++mt)
#pragma unroll
    for (int nt = 0; nt < 4; ++nt)
#pragma unroll
      for (int r = 0; r < 8; ++r) {
        const int m = m0 + wr + mt * 16 + ((lane >> 4) << 3) + r;
        const int c = n0 + wc + nt * 16 + (lane & 15);
        const int which = c >> 10;
        const int cc = c & 1023;
        const int bb = m >> 11, tt = m & 2047;
        const int hh = cc >> 6, dd = cc & 63;
        const float val = acc[mt][nt][r];
        const size_t hidx = (((size_t)bb * H_ + hh) * T_ + tt) * HD_ + dd;
        if (which == 0) {
          Qbf[hidx] = f2bf(val);
        } else if (which == 1) {
          Kout[hidx] = val;
          Kbf[hidx] = f2bf(val);
        } else {
          Vout[hidx] = val;
          Vtbf[(((size_t)bb * H_ + hh) * HD_ + dd) * T_ + tt] = f2bf(val);
        }
      }
}

// ---------------------------------------------------------------------------
// Kernel 2: flash attention, double-buffered TDM K/V prefetch, parallel softmax.
// grid = (B*H, T/64).
// ---------------------------------------------------------------------------
__global__ __launch_bounds__(256) void attn_tdm(
    const __bf16* __restrict__ Qbf, const __bf16* __restrict__ Kbf,
    const __bf16* __restrict__ Vtbf, __bf16* __restrict__ Obf) {
  constexpr int LD = 66;
  __shared__ __bf16 Qs[64 * LD];
  __shared__ __bf16 Ks[2][64 * LD];    // [key][d]
  __shared__ __bf16 Vts[2][64 * LD];   // [d][key]
  __shared__ float  Sb[64 * LD];
  __shared__ __bf16 Pb[64 * LD];
  __shared__ float  pmax[64][4];
  __shared__ float  psum[64][4];
  __shared__ float  mrow[64];
  __shared__ float  alphas[64];
  __shared__ float  invl[64];

  const int tid = threadIdx.x, lane = tid & 31, w = tid >> 5;
  const int bh = blockIdx.x;
  const int q0 = blockIdx.y * 64;
  const int rbase = (w & 3) * 16;
  const int cbase = (w >> 2) * 32;
  const int srow = tid >> 2;           // softmax: 4 threads per row
  const int sseg = (tid & 3) << 4;     // 16-column segment

  if (w == 0) {
    tdm_load_2d(lds_off(&Qs[0]),     Qbf  + ((size_t)bh * T_ + q0) * HD_, 64, 64, HD_, 4);
    tdm_load_2d(lds_off(&Ks[0][0]),  Kbf  + ((size_t)bh * T_) * HD_,      64, 64, HD_, 4);
    tdm_load_2d(lds_off(&Vts[0][0]), Vtbf + ((size_t)bh * HD_) * T_,      64, 64, T_,  4);
  }

  const v8f vz = {0.f, 0.f, 0.f, 0.f, 0.f, 0.f, 0.f, 0.f};
  v8f o0 = vz, o1 = vz;
  float m_run = -3.0e38f, l_run = 0.0f;

  const int nkb = blockIdx.y + 1;
  for (int kb = 0; kb < nkb; ++kb) {
    const int cur = kb & 1;
    if (kb + 1 < nkb) {
      if (w == 0) {
        tdm_load_2d(lds_off(&Ks[cur ^ 1][0]),
                    Kbf + ((size_t)bh * T_ + (size_t)(kb + 1) * 64) * HD_, 64, 64, HD_, 4);
        tdm_load_2d(lds_off(&Vts[cur ^ 1][0]),
                    Vtbf + ((size_t)bh * HD_) * T_ + (size_t)(kb + 1) * 64, 64, 64, T_, 4);
      }
      __builtin_amdgcn_s_wait_tensorcnt(2);   // current K/V (and Q) complete
    } else {
      __builtin_amdgcn_s_wait_tensorcnt(0);
    }
    __syncthreads();

    // S = Q K^T
    v8f s0 = vz, s1 = vz;
#pragma unroll
    for (int kc = 0; kc < 2; ++kc) {
      v16bf aq = frag_a16x32(Qs, rbase, kc * 32, LD);
      v16bf b0 = frag_b32x16(&Ks[cur][0], cbase,      kc * 32, LD);
      v16bf b1 = frag_b32x16(&Ks[cur][0], cbase + 16, kc * 32, LD);
      s0 = wmma_bf16(aq, b0, s0);
      s1 = wmma_bf16(aq, b1, s1);
    }
#pragma unroll
    for (int r = 0; r < 8; ++r) {
      const int row = rbase + ((lane >> 4) << 3) + r;
      Sb[row * LD + cbase + (lane & 15)]      = s0[r];
      Sb[row * LD + cbase + 16 + (lane & 15)] = s1[r];
    }
    __syncthreads();

    // ---- parallel online softmax (1/sqrt(hd)=0.125 folded in) ----
    int jmax = (q0 + srow) - kb * 64 + 1;    // causal limit inside block
    if (jmax > 64) jmax = 64;
    {
      float pm = -3.0e38f;
      const float* sr = &Sb[srow * LD];
      const int je = (jmax < sseg + 16) ? jmax : (sseg + 16);
      for (int j = sseg; j < je; ++j) pm = fmaxf(pm, sr[j] * 0.125f);
      pmax[srow][tid & 3] = pm;
    }
    __syncthreads();
    if (tid < 64) {
      const float mloc = fmaxf(fmaxf(pmax[tid][0], pmax[tid][1]),
                               fmaxf(pmax[tid][2], pmax[tid][3]));
      const float m_new = fmaxf(m_run, mloc);
      alphas[tid] = (m_run <= -1.0e38f) ? 0.0f : __expf(m_run - m_new);
      mrow[tid] = m_new;
      m_run = m_new;
    }
    __syncthreads();
    {
      const float mn = mrow[srow];
      const float* sr = &Sb[srow * LD];
      __bf16* pr = &Pb[srow * LD];
      float ps = 0.0f;
      const int je = (jmax < sseg + 16) ? jmax : (sseg + 16);
      for (int j = sseg; j < sseg + 16; ++j) {
        if (j < je) {
          const float p = __expf(sr[j] * 0.125f - mn);
          ps += p;
          pr[j] = f2bf(p);
        } else {
          pr[j] = f2bf(0.0f);
        }
      }
      psum[srow][tid & 3] = ps;
    }
    __syncthreads();
    if (tid < 64)
      l_run = l_run * alphas[tid] + psum[tid][0] + psum[tid][1] + psum[tid][2] + psum[tid][3];

    // ---- O = O*alpha + P V ----
#pragma unroll
    for (int r = 0; r < 8; ++r) {
      const float a = alphas[rbase + ((lane >> 4) << 3) + r];
      o0[r] *= a;
      o1[r] *= a;
    }
#pragma unroll
    for (int kc = 0; kc < 2; ++kc) {
      v16bf ap = frag_a16x32(Pb, rbase, kc * 32, LD);
      v16bf b0 = frag_b32x16(&Vts[cur][0], cbase,      kc * 32, LD);
      v16bf b1 = frag_b32x16(&Vts[cur][0], cbase + 16, kc * 32, LD);
      o0 = wmma_bf16(ap, b0, o0);
      o1 = wmma_bf16(ap, b1, o1);
    }
    __syncthreads();   // done with Pb and buffers before next TDM overwrite
  }

  if (tid < 64) invl[tid] = 1.0f / l_run;
  __syncthreads();

  const int bb = bh >> 4, hh = bh & 15;
#pragma unroll
  for (int r = 0; r < 8; ++r) {
    const int lrow = rbase + ((lane >> 4) << 3) + r;
    const int t = q0 + lrow;
    const float il = invl[lrow];
    const size_t rowoff = ((size_t)bb * T_ + t) * C_ + (size_t)hh * HD_;
    Obf[rowoff + cbase + (lane & 15)]      = f2bf(o0[r] * il);
    Obf[rowoff + cbase + 16 + (lane & 15)] = f2bf(o1[r] * il);
  }
}

// ---------------------------------------------------------------------------
// Kernel 3: out = O @ W_proj + b_proj  (k-step 64, double-buffered TDM).
// ---------------------------------------------------------------------------
__global__ __launch_bounds__(256) void proj_gemm_tdm(
    const __bf16* __restrict__ Abf, const __bf16* __restrict__ Btbf,
    const float* __restrict__ bias, float* __restrict__ out) {
  constexpr int LD = 66;
  __shared__ __bf16 As[2][128 * LD];
  __shared__ __bf16 Bts[2][128 * LD];
  const int tid = threadIdx.x, lane = tid & 31, w = tid >> 5;
  const int m0 = blockIdx.y * 128, n0 = blockIdx.x * 128;
  const int wr = (w & 3) * 32, wc = (w >> 2) * 64;
  const v8f vz = {0.f, 0.f, 0.f, 0.f, 0.f, 0.f, 0.f, 0.f};
  v8f acc[2][4];
#pragma unroll
  for (int mt = 0; mt < 2; ++mt)
#pragma unroll
    for (int nt = 0; nt < 4; ++nt) acc[mt][nt] = vz;

  if (w == 0) {
    tdm_load_2d(lds_off(&As[0][0]),  Abf  + (size_t)m0 * C_, 64, 128, C_, 4);
    tdm_load_2d(lds_off(&Bts[0][0]), Btbf + (size_t)n0 * C_, 64, 128, C_, 4);
  }
  for (int it = 0; it < 16; ++it) {
    const int cur = it & 1;
    if (it + 1 < 16) {
      if (w == 0) {
        tdm_load_2d(lds_off(&As[cur ^ 1][0]),  Abf  + (size_t)m0 * C_ + (it + 1) * 64, 64, 128, C_, 4);
        tdm_load_2d(lds_off(&Bts[cur ^ 1][0]), Btbf + (size_t)n0 * C_ + (it + 1) * 64, 64, 128, C_, 4);
      }
      __builtin_amdgcn_s_wait_tensorcnt(2);
    } else {
      __builtin_amdgcn_s_wait_tensorcnt(0);
    }
    __syncthreads();
#pragma unroll
    for (int kc = 0; kc < 2; ++kc) {
      v16bf a0 = frag_a16x32(&As[cur][0], wr, kc * 32, LD);
      v16bf a1 = frag_a16x32(&As[cur][0], wr + 16, kc * 32, LD);
#pragma unroll
      for (int nt = 0; nt < 4; ++nt) {
        v16bf bfr = frag_b32x16(&Bts[cur][0], wc + nt * 16, kc * 32, LD);
        acc[0][nt] = wmma_bf16(a0, bfr, acc[0][nt]);
        acc[1][nt] = wmma_bf16(a1, bfr, acc[1][nt]);
      }
    }
    __syncthreads();
  }

#pragma unroll
  for (int mt = 0; mt < 2; ++mt)
#pragma unroll
    for (int nt = 0; nt < 4; ++nt)
#pragma unroll
      for (int r = 0; r < 8; ++r) {
        const int m = m0 + wr + mt * 16 + ((lane >> 4) << 3) + r;
        const int c = n0 + wc + nt * 16 + (lane & 15);
        out[(size_t)m * C_ + c] = acc[mt][nt][r] + bias[c];
      }
}

// ---------------------------------------------------------------------------
extern "C" void kernel_launch(void* const* d_in, const int* in_sizes, int n_in,
                              void* d_out, int out_size, void* d_ws, size_t ws_size,
                              hipStream_t stream) {
  (void)in_sizes; (void)n_in; (void)out_size; (void)ws_size;
  const float* x     = (const float*)d_in[0];
  const float* Wqkv  = (const float*)d_in[1];
  const float* Wproj = (const float*)d_in[2];
  const float* bproj = (const float*)d_in[3];

  float* out = (float*)d_out;
  const size_t OE = (size_t)M_ * C_;
  float* Kout = out + OE;
  float* Vout = out + 2 * OE;

  // Workspace carve (bf16 buffers), ~88 MB total.
  char* ws = (char*)d_ws;
  __bf16* Xbf    = (__bf16*)ws; ws += (size_t)M_ * C_ * 2;
  __bf16* Wqkvt  = (__bf16*)ws; ws += (size_t)NQKV * C_ * 2;
  __bf16* Wprojt = (__bf16*)ws; ws += (size_t)C_ * C_ * 2;
  __bf16* Qbf    = (__bf16*)ws; ws += (size_t)B_ * H_ * T_ * HD_ * 2;
  __bf16* Kbf    = (__bf16*)ws; ws += (size_t)B_ * H_ * T_ * HD_ * 2;
  __bf16* Vtbf   = (__bf16*)ws; ws += (size_t)B_ * H_ * T_ * HD_ * 2;
  __bf16* Obf    = (__bf16*)ws; ws += (size_t)M_ * C_ * 2;

  dim3 blk(256, 1, 1);
  pack_bf16_k<<<dim3((M_ * C_ / 4 + 255) / 256), blk, 0, stream>>>(x, Xbf, M_ * C_ / 4);
  transpose_pack_k<<<dim3((C_ * NQKV + 255) / 256), blk, 0, stream>>>(Wqkv, Wqkvt, C_, NQKV);
  transpose_pack_k<<<dim3((C_ * C_ + 255) / 256), blk, 0, stream>>>(Wproj, Wprojt, C_, C_);

  qkv_gemm_tdm<<<dim3(NQKV / 128, M_ / 128, 1), blk, 0, stream>>>(
      Xbf, Wqkvt, Kout, Vout, Qbf, Kbf, Vtbf);
  attn_tdm<<<dim3(B_ * H_, T_ / 64, 1), blk, 0, stream>>>(Qbf, Kbf, Vtbf, Obf);
  proj_gemm_tdm<<<dim3(C_ / 128, M_ / 128, 1), blk, 0, stream>>>(Obf, Wprojt, bproj, out);
}
